// GPT_63788854280955
// MI455X (gfx1250) — matrix operand order repaired and checked by reading, not compiled, not run on hardware
//
#include <hip/hip_runtime.h>
#include <hip/hip_bf16.h>
#include <math.h>

typedef __attribute__((ext_vector_type(16))) _Float16 v16h;
typedef __attribute__((ext_vector_type(8)))  _Float16 v8h;
typedef __attribute__((ext_vector_type(8)))  float    v8f;

#define GB   4
#define GT   2048
#define GV   99
#define GD   512
#define GH   8
#define GDH  64
#define GL   6
#define NTOK (GB * GT)          // 8192 tokens
#define OFFS 28                 // banded-causal lookahead
#define GELU_C 0.7978845608028654f

// Ensure same-wave LDS write->read ordering without a block barrier
// (waves in a block have divergent loop counts in the attention kernel).
#define WAVE_LDS_SYNC() do {                                    \
    __builtin_amdgcn_wave_barrier();                            \
    asm volatile("s_wait_dscnt 0" ::: "memory");                \
    __builtin_amdgcn_wave_barrier();                            \
} while (0)

// ---------------- WMMA fragment loaders (16x16x32 f16, contiguous k = 16*hi + s order) --------
// A 16x32: lane (m=l%16, hi=l/16) holds K = [16*hi, 16*hi+16)
__device__ __forceinline__ v16h load_a_frag(const _Float16* rowp, int k0, int hi) {
    union { v16h v; v8h h[2]; } u;
    u.h[0] = *(const v8h*)(rowp + k0 + 16 * hi);
    u.h[1] = *(const v8h*)(rowp + k0 + 16 * hi + 8);
    return u.v;
}
// B 32x16: lane (n=l%16, hi=l/16) holds K = [16*hi, 16*hi+16) of column n.
// Source is [N,K] row-major -> one contiguous 32B read of row n.
__device__ __forceinline__ v16h load_b_frag(const _Float16* rowp, int k0, int hi) {
    return *(const v16h*)(rowp + k0 + 16 * hi);
}

// ---------------- Epilogues ------------------------------------------------
struct EpiQKV {   // scatter qkv: q,k -> [B*H, T, DH], v -> transposed [B*H, DH, T]
    _Float16 *q, *k, *vt;
    __device__ void operator()(int m, int n, float v) const {
        int b = m >> 11, t = m & (GT - 1);
        int which = n >> 9, o = n & (GD - 1);
        int h = o >> 6, dh = o & 63;
        size_t bh = (size_t)(b * GH + h);
        _Float16 hv = (_Float16)v;
        if (which == 0)      q[(bh * GT + t) * GDH + dh] = hv;
        else if (which == 1) k[(bh * GT + t) * GDH + dh] = hv;
        else                 vt[(bh * GDH + dh) * GT + t] = hv;
    }
};
struct EpiAddRes { // residual add into fp32 activations
    float* x;
    __device__ void operator()(int m, int n, float v) const {
        x[(size_t)m * GD + n] += v;
    }
};
struct EpiGelu {   // new_gelu -> f16 MLP intermediate [NTOK, 4D]
    _Float16* o;
    __device__ void operator()(int m, int n, float v) const {
        float g = 0.5f * v * (1.f + tanhf(GELU_C * (v + 0.044715f * v * v * v)));
        o[(size_t)m * (4 * GD) + n] = (_Float16)g;
    }
};
struct EpiLogits { // fp32 logits straight to d_out
    float* o;
    __device__ void operator()(int m, int n, float v) const {
        o[(size_t)m * GV + n] = v;
    }
};

// ---------------- Fused WMMA GEMM: C[M,N] = A[M,K] * W[N,K]^T --------------
// One wave computes a 32x64 tile: 2 M-subtiles x 4 N-subtiles. Each B frag is
// reused by 2 WMMAs, each A frag by 4 -> 24 bytes loaded per lane per WMMA.
template <class Epi>
__global__ __launch_bounds__(256) void gemm_f16_wmma(
    const _Float16* __restrict__ A, const _Float16* __restrict__ Bw,
    int M, int N, int K, Epi epi)
{
    const int ntile = (N + 63) >> 6;
    const int wave  = blockIdx.x * (blockDim.x >> 5) + (threadIdx.x >> 5);
    const int mt = wave / ntile;
    const int nt = wave % ntile;
    if (mt * 32 >= M) return;                    // wave-uniform: EXEC stays full
    const int lane = threadIdx.x & 31, ln = lane & 15, hi = lane >> 4;

    const _Float16* arow0 = A + (size_t)(mt * 32 + ln) * K;
    const _Float16* arow1 = A + (size_t)(mt * 32 + 16 + ln) * K;
    const _Float16* brow[4];
#pragma unroll
    for (int s = 0; s < 4; ++s) {
        int n = nt * 64 + s * 16 + ln;
        brow[s] = Bw + (size_t)min(n, N - 1) * K;   // clamp for N=99 edge tile
    }
    v8f acc0[4] = {v8f{}, v8f{}, v8f{}, v8f{}};
    v8f acc1[4] = {v8f{}, v8f{}, v8f{}, v8f{}};
    for (int k0 = 0; k0 < K; k0 += 32) {
        v16h a0 = load_a_frag(arow0, k0, hi);
        v16h a1 = load_a_frag(arow1, k0, hi);
#pragma unroll
        for (int s = 0; s < 4; ++s) {
            v16h b = load_b_frag(brow[s], k0, hi);
            acc0[s] = __builtin_amdgcn_wmma_f32_16x16x32_f16(
                false, a0, false, b, (short)0, acc0[s], false, false);
            acc1[s] = __builtin_amdgcn_wmma_f32_16x16x32_f16(
                false, a1, false, b, (short)0, acc1[s], false, false);
        }
    }
#pragma unroll
    for (int s = 0; s < 4; ++s) {
        int n = nt * 64 + s * 16 + ln;
        if (n < N) {
#pragma unroll
            for (int r = 0; r < 8; ++r) {
                epi(mt * 32 + r + 8 * hi, n, acc0[s][r]);
                epi(mt * 32 + 16 + r + 8 * hi, n, acc1[s][r]);
            }
        }
    }
}

// ---------------- Flash attention (banded mask j <= i+28) ------------------
// One wave per (batch*head, 16-query tile). Processes keys 32 at a time.
__global__ __launch_bounds__(256) void attn_wmma(
    const _Float16* __restrict__ Q, const _Float16* __restrict__ Kb,
    const _Float16* __restrict__ Vt, _Float16* __restrict__ Y)
{
    __shared__ _Float16 Plds[8][16][32];
    const int wid = threadIdx.x >> 5, lane = threadIdx.x & 31;
    const int ln = lane & 15, hi = lane >> 4;
    const int gw = blockIdx.x * 8 + wid;
    const int qt = gw & (GT / 16 - 1);
    const int bh = gw >> 7;
    const _Float16* Qp = Q  + (size_t)bh * GT * GDH;
    const _Float16* Kp = Kb + (size_t)bh * GT * GDH;
    const _Float16* Vp = Vt + (size_t)bh * GDH * GT;
    const int qb = qt * 16;

    const _Float16* qrow = Qp + (size_t)(qb + ln) * GDH;
    v16h qa0 = load_a_frag(qrow, 0, hi);
    v16h qa1 = load_a_frag(qrow, 32, hi);

    v8f acc[4] = {v8f{}, v8f{}, v8f{}, v8f{}};
    float mrow[8], lrow[8];
#pragma unroll
    for (int r = 0; r < 8; ++r) { mrow[r] = -1e30f; lrow[r] = 0.f; }

    const float scale = 0.125f;                         // 1/sqrt(64)
    const int kend = min(qb + 15 + OFFS, GT - 1);
    for (int kb = 0; kb <= kend; kb += 32) {
        v8f S[2];
#pragma unroll
        for (int s = 0; s < 2; ++s) {
            int krow = min(kb + s * 16 + ln, GT - 1);
            const _Float16* kr = Kp + (size_t)krow * GDH;
            v8f z = {};
            z = __builtin_amdgcn_wmma_f32_16x16x32_f16(
                false, qa0, false, load_b_frag(kr, 0, hi), (short)0, z, false, false);
            z = __builtin_amdgcn_wmma_f32_16x16x32_f16(
                false, qa1, false, load_b_frag(kr, 32, hi), (short)0, z, false, false);
            S[s] = z;
        }
        float newm[8];
#pragma unroll
        for (int r = 0; r < 8; ++r) {
            int q = qb + r + 8 * hi;
            float v = -1e30f;
#pragma unroll
            for (int s = 0; s < 2; ++s) {
                int k = kb + s * 16 + ln;
                float sv = (k <= q + OFFS && k < GT) ? S[s][r] * scale : -1e30f;
                S[s][r] = sv;
                v = fmaxf(v, sv);
            }
#pragma unroll
            for (int off = 8; off >= 1; off >>= 1)
                v = fmaxf(v, __shfl_xor(v, off, 32));   // row lives in a 16-lane half
            newm[r] = fmaxf(mrow[r], v);
        }
        WAVE_LDS_SYNC();   // prior iteration's P reads done before overwrite
        float rsum[8];
#pragma unroll
        for (int r = 0; r < 8; ++r) {
            float rs = 0.f;
#pragma unroll
            for (int s = 0; s < 2; ++s) {
                float p = __expf(S[s][r] - newm[r]);
                rs += p;
                Plds[wid][r + 8 * hi][s * 16 + ln] = (_Float16)p;
            }
#pragma unroll
            for (int off = 8; off >= 1; off >>= 1)
                rs += __shfl_xor(rs, off, 32);
            rsum[r] = rs;
        }
#pragma unroll
        for (int r = 0; r < 8; ++r) {
            float f = __expf(mrow[r] - newm[r]);
            lrow[r] = lrow[r] * f + rsum[r];
            mrow[r] = newm[r];
#pragma unroll
            for (int t = 0; t < 4; ++t) acc[t][r] *= f;
        }
        WAVE_LDS_SYNC();   // P stores visible before A-frag reads
        v16h pa = load_a_frag(&Plds[wid][ln][0], 0, hi);
        const int vcol = min(kb + 16 * hi, GT - 16);
#pragma unroll
        for (int t = 0; t < 4; ++t) {
            const _Float16* vrow = Vp + (size_t)(t * 16 + ln) * GT;
            v16h vb = *(const v16h*)(vrow + vcol);
            acc[t] = __builtin_amdgcn_wmma_f32_16x16x32_f16(
                false, pa, false, vb, (short)0, acc[t], false, false);
        }
    }
    const int b = bh >> 3, h = bh & 7;
#pragma unroll
    for (int r = 0; r < 8; ++r) {
        float inv = 1.f / lrow[r];
        int tok = b * GT + qb + r + 8 * hi;
#pragma unroll
        for (int t = 0; t < 4; ++t)
            Y[(size_t)tok * GD + h * GDH + t * 16 + ln] = (_Float16)(acc[t][r] * inv);
    }
}

// ---------------- LayerNorm: one wave per row, f16 output ------------------
__global__ __launch_bounds__(256) void layernorm_f16(
    const float* __restrict__ X, const float* __restrict__ w,
    _Float16* __restrict__ O, int M)
{
    int wave = blockIdx.x * 8 + (threadIdx.x >> 5);
    if (wave >= M) return;
    int lane = threadIdx.x & 31;
    const float* row = X + (size_t)wave * GD;
    float v[16]; float s = 0.f;
#pragma unroll
    for (int i = 0; i < 16; ++i) { v[i] = row[lane * 16 + i]; s += v[i]; }
#pragma unroll
    for (int off = 16; off >= 1; off >>= 1) s += __shfl_xor(s, off, 32);
    float mean = s * (1.f / GD);
    float var = 0.f;
#pragma unroll
    for (int i = 0; i < 16; ++i) { float d = v[i] - mean; var += d * d; }
#pragma unroll
    for (int off = 16; off >= 1; off >>= 1) var += __shfl_xor(var, off, 32);
    float rstd = rsqrtf(var * (1.f / GD) + 1e-5f);
    _Float16* orow = O + (size_t)wave * GD;
#pragma unroll
    for (int i = 0; i < 16; ++i)
        orow[lane * 16 + i] = (_Float16)((v[i] - mean) * rstd * w[lane * 16 + i]);
}

// ---------------- Elementwise helpers --------------------------------------
__global__ void embed_k(const int* __restrict__ idx, const float* __restrict__ wte,
                        const float* __restrict__ wpe, float* __restrict__ X)
{
    size_t i = (size_t)blockIdx.x * blockDim.x + threadIdx.x;
    if (i >= (size_t)NTOK * GD) return;
    int d = (int)(i & (GD - 1));
    int tok = (int)(i >> 9);
    int t = tok & (GT - 1);
    X[i] = wte[(size_t)idx[tok] * GD + d] + wpe[(size_t)t * GD + d];
}

__global__ void cvt_f16_k(const float* __restrict__ in, _Float16* __restrict__ out, size_t n)
{
    size_t i = (size_t)blockIdx.x * blockDim.x + threadIdx.x;
    if (i < n) out[i] = (_Float16)in[i];
}

// ---------------- Weighted CE loss (deterministic two-stage) ---------------
__global__ __launch_bounds__(256) void loss_partial_k(
    const float* __restrict__ logits, const int* __restrict__ tgt,
    float* __restrict__ partial)
{
    __shared__ float sA[8], sB[8];
    int wid = threadIdx.x >> 5, lane = threadIdx.x & 31;
    int token = blockIdx.x * 8 + wid;
    const float* row = logits + (size_t)token * GV;
    float mx = -1e30f;
#pragma unroll
    for (int j = 0; j < 4; ++j) { int i = lane + 32 * j; if (i < GV) mx = fmaxf(mx, row[i]); }
#pragma unroll
    for (int off = 16; off >= 1; off >>= 1) mx = fmaxf(mx, __shfl_xor(mx, off, 32));
    float se = 0.f;
#pragma unroll
    for (int j = 0; j < 4; ++j) { int i = lane + 32 * j; if (i < GV) se += __expf(row[i] - mx); }
#pragma unroll
    for (int off = 16; off >= 1; off >>= 1) se += __shfl_xor(se, off, 32);
    int y = tgt[token];
    int valid = (y != -1);
    int yc = valid ? y : 0;
    float w = (yc >= 78 && yc < 96) ? 1.f : (yc == 96 ? 0.1f : 0.f);
    float wi = valid ? w : 0.f;
    float nll = -(row[yc] - mx - __logf(se));
    if (lane == 0) { sA[wid] = wi * nll; sB[wid] = wi; }
    __syncthreads();
    if (threadIdx.x == 0) {
        float a = 0.f, b = 0.f;
        for (int i = 0; i < 8; ++i) { a += sA[i]; b += sB[i]; }
        partial[blockIdx.x * 2] = a; partial[blockIdx.x * 2 + 1] = b;
    }
}

__global__ void loss_final_k(const float* __restrict__ partial, float* __restrict__ loss, int nblk)
{
    if (blockIdx.x == 0 && threadIdx.x == 0) {
        float a = 0.f, b = 0.f;
        for (int i = 0; i < nblk; ++i) { a += partial[2 * i]; b += partial[2 * i + 1]; }
        loss[0] = a / b;
    }
}

// ---------------- Host orchestration ---------------------------------------
extern "C" void kernel_launch(void* const* d_in, const int* in_sizes, int n_in,
                              void* d_out, int out_size, void* d_ws, size_t ws_size,
                              hipStream_t stream)
{
    (void)in_sizes; (void)n_in; (void)out_size; (void)ws_size;
    const int*   idx       = (const int*)d_in[0];
    const int*   targets   = (const int*)d_in[1];
    const float* wte       = (const float*)d_in[2];
    const float* wpe       = (const float*)d_in[3];
    const float* ln1_w     = (const float*)d_in[4];
    const float* attn_w    = (const float*)d_in[5];
    const float* proj_w    = (const float*)d_in[6];
    const float* ln2_w     = (const float*)d_in[7];
    const float* fc_w      = (const float*)d_in[8];
    const float* fc_proj_w = (const float*)d_in[9];
    const float* lnf_w     = (const float*)d_in[10];

    char* ws = (char*)d_ws;
    size_t off = 0;
    auto alloc = [&](size_t bytes) -> char* {
        char* p = ws + off;
        off = (off + bytes + 255) & ~(size_t)255;
        return p;
    };
    float*     x     = (float*)    alloc((size_t)NTOK * GD * 4);
    _Float16*  h16   = (_Float16*) alloc((size_t)NTOK * GD * 2);
    _Float16*  q16   = (_Float16*) alloc((size_t)NTOK * GD * 2);
    _Float16*  k16   = (_Float16*) alloc((size_t)NTOK * GD * 2);
    _Float16*  vt16  = (_Float16*) alloc((size_t)NTOK * GD * 2);
    _Float16*  y16   = (_Float16*) alloc((size_t)NTOK * GD * 2);
    _Float16*  m16   = (_Float16*) alloc((size_t)NTOK * 4 * GD * 2);
    _Float16*  wa16  = (_Float16*) alloc((size_t)GL * 3 * GD * GD * 2);
    _Float16*  wp16  = (_Float16*) alloc((size_t)GL * GD * GD * 2);
    _Float16*  wf16  = (_Float16*) alloc((size_t)GL * 4 * GD * GD * 2);
    _Float16*  wfp16 = (_Float16*) alloc((size_t)GL * GD * 4 * GD * 2);
    _Float16*  wte16 = (_Float16*) alloc((size_t)GV * GD * 2);
    float*     part  = (float*)    alloc((size_t)(NTOK / 8) * 2 * 4);

    auto cvt = [&](const float* src, _Float16* dst, size_t n) {
        unsigned blocks = (unsigned)((n + 255) / 256);
        cvt_f16_k<<<blocks, 256, 0, stream>>>(src, dst, n);
    };
    cvt(attn_w,    wa16,  (size_t)GL * 3 * GD * GD);
    cvt(proj_w,    wp16,  (size_t)GL * GD * GD);
    cvt(fc_w,      wf16,  (size_t)GL * 4 * GD * GD);
    cvt(fc_proj_w, wfp16, (size_t)GL * GD * 4 * GD);
    cvt(wte,       wte16, (size_t)GV * GD);

    embed_k<<<(NTOK * GD + 255) / 256, 256, 0, stream>>>(idx, wte, wpe, x);

    auto gemm = [&](const _Float16* A, const _Float16* Bw, int M, int N, int K, auto epi) {
        int ntile = (N + 63) >> 6;
        int waves = (M >> 5) * ntile;        // 32-row tiles
        int blocks = (waves + 7) >> 3;
        gemm_f16_wmma<<<blocks, 256, 0, stream>>>(A, Bw, M, N, K, epi);
    };

    for (int l = 0; l < GL; ++l) {
        layernorm_f16<<<NTOK / 8, 256, 0, stream>>>(x, ln1_w + (size_t)l * GD, h16, NTOK);
        gemm(h16, wa16 + (size_t)l * 3 * GD * GD, NTOK, 3 * GD, GD, EpiQKV{q16, k16, vt16});
        attn_wmma<<<(GB * GH * (GT / 16)) / 8, 256, 0, stream>>>(q16, k16, vt16, y16);
        gemm(y16, wp16 + (size_t)l * GD * GD, NTOK, GD, GD, EpiAddRes{x});
        layernorm_f16<<<NTOK / 8, 256, 0, stream>>>(x, ln2_w + (size_t)l * GD, h16, NTOK);
        gemm(h16, wf16 + (size_t)l * 4 * GD * GD, NTOK, 4 * GD, GD, EpiGelu{m16});
        gemm(m16, wfp16 + (size_t)l * GD * 4 * GD, NTOK, GD, 4 * GD, EpiAddRes{x});
    }

    layernorm_f16<<<NTOK / 8, 256, 0, stream>>>(x, lnf_w, h16, NTOK);
    float* logits = (float*)d_out;
    gemm(h16, wte16, NTOK, GV, GD, EpiLogits{logits});

    loss_partial_k<<<NTOK / 8, 256, 0, stream>>>(logits, targets, part);
    loss_final_k<<<1, 32, 0, stream>>>(part, logits + (size_t)NTOK * GV, NTOK / 8);
}
